// ObjectDetector_5196910428632
// MI455X (gfx1250) — compile-verified, matrix operand order checked by reading, not run on hardware
//
#include <hip/hip_runtime.h>

// ---------------------------------------------------------------------------
// Types
// ---------------------------------------------------------------------------
typedef _Float16 h16;
typedef __attribute__((ext_vector_type(16))) _Float16 v16h;
typedef __attribute__((ext_vector_type(8)))  float    v8f;
typedef __attribute__((ext_vector_type(4)))  unsigned int u32x4;
typedef __attribute__((ext_vector_type(4)))  int i32x4;
typedef __attribute__((ext_vector_type(8)))  int i32x8;

// ---------------------------------------------------------------------------
// Problem constants (from reference)
// ---------------------------------------------------------------------------
#define IMG      1024
#define FB       256              // backbone output spatial
#define C1       128              // backbone channels
#define FH       254              // inter conv output spatial
#define C2       256              // inter conv channels
#define NPIX     (FH * FH)        // 64516
#define NANCH    (NPIX * 9)       // 580644
#define KGT      16
#define NCOMB    48               // 9 cls + 36 reg + 3 pad
#define KI       1152             // 3*3*128 inter-conv K
#define ABLK     2269             // ceil(NANCH/256)

#if defined(__AMDGCN__) && __has_builtin(__builtin_amdgcn_tensor_load_to_lds)
#define HAVE_TDM 1
#else
#define HAVE_TDM 0
#endif

// ---------------------------------------------------------------------------
// WMMA fragment helpers (CDNA5 ISA 7.12.2, wave32).
// 16-bit A (16x32 MxK) / B^T (N-major, K contiguous):
//   lane row r = lane&15, kh = (lane>>4)*8
//   halves 0-7  = k: kh..kh+7 ; halves 8-15 = k: 16+kh..23+kh
// -> per fragment: two contiguous 16-byte loads from a K-major row.
// ---------------------------------------------------------------------------
__device__ __forceinline__ v16h frag_ld(const h16* p0) {
  v16h r;
  ((uint4*)&r)[0] = *(const uint4*)(p0);        // k: kh..kh+7
  ((uint4*)&r)[1] = *(const uint4*)(p0 + 16);   // k: 16+kh..23+kh
  return r;
}

__device__ __forceinline__ v16h frag_kmajor(const h16* src, int ld, int lane) {
  int r = lane & 15, kh = (lane >> 4) << 3;
  return frag_ld(src + r * ld + kh);
}

__device__ __forceinline__ v8f wmma32(v16h a, v16h b, v8f c) {
  return __builtin_amdgcn_wmma_f32_16x16x32_f16(false, a, false, b,
                                                (short)0, c, false, false);
}

// ---------------------------------------------------------------------------
// Ordered float <-> uint key (monotonic) for deterministic float max-reduce
// ---------------------------------------------------------------------------
__device__ __forceinline__ unsigned int fkey(float f) {
  unsigned int b = __float_as_uint(f);
  return (b & 0x80000000u) ? ~b : (b | 0x80000000u);
}
__device__ __forceinline__ float funkey(unsigned int k) {
  unsigned int b = (k & 0x80000000u) ? (k & 0x7fffffffu) : ~k;
  return __uint_as_float(b);
}

__device__ __forceinline__ float softplusf(float x) {
  return fmaxf(x, 0.f) + log1pf(expf(-fabsf(x)));
}

__device__ __forceinline__ void anchor_geom(int a, float& hw, float& hh) {
  const float ratios[3] = {0.5f, 1.0f, 2.0f};
  const float scales[3] = {0.2f, 0.4f, 0.7f};
  float rs = sqrtf(ratios[a / 3]);
  float sc = scales[a % 3];
  hw = 0.5f * sc * rs;
  hh = 0.5f * sc / rs;
}

// ---------------------------------------------------------------------------
// K0: build K-major (transposed) f16 weights + init gt-max keys
// ---------------------------------------------------------------------------
__global__ void __launch_bounds__(256)
k_prep(const float* __restrict__ Wb, const float* __restrict__ Wi,
       const float* __restrict__ Wc, const float* __restrict__ Wr,
       h16* __restrict__ WbT, h16* __restrict__ WihT, h16* __restrict__ WcrT,
       unsigned int* __restrict__ gtkey) {
  int t = blockIdx.x * 256 + threadIdx.x;
  if (t < 8192) {
    int n = t >> 6, k = t & 63;
    WbT[t] = (k < 48) ? (h16)Wb[k * C1 + n] : (h16)0.f;
    return;
  }
  t -= 8192;
  if (t < C2 * KI) {
    int n = t / KI, k = t % KI;
    WihT[t] = (h16)Wi[k * C2 + n];
    return;
  }
  t -= C2 * KI;
  if (t < NCOMB * C2) {
    int n = t >> 8, k = t & 255;
    float v = (n < 9) ? Wc[k * 9 + n] : (n < 45 ? Wr[k * 36 + (n - 9)] : 0.f);
    WcrT[t] = (h16)v;
    return;
  }
  t -= NCOMB * C2;
  if (t < KGT) gtkey[t] = 0u;
}

// ---------------------------------------------------------------------------
// K1: backbone conv 4x4 stride4, 3->128.
// Each lane needs exactly 24 floats of its im2col row; for either kh half
// they form six float4-aligned runs (k in {kh, kh+4, 16+kh, 20+kh, 32+kh,
// 36+kh}; ky=k/12, r=k%12, r in {0,4,8}).  Addresses are computed from kh
// (no data-selects -> no scratch spill); fragments packed with constant
// element indices.  Two K=32 WMMA steps (K=48 padded to 64).
// ---------------------------------------------------------------------------
__global__ void __launch_bounds__(256)
k_backbone(const float* __restrict__ img, const h16* __restrict__ WbT,
           const float* __restrict__ bb, h16* __restrict__ feats) {
  int lane = threadIdx.x & 31;
  int wid = blockIdx.x * 8 + (threadIdx.x >> 5);   // 0..4095
  int y = wid >> 4;                                 // 0..255
  int x0 = (wid & 15) << 4;
  int m = lane & 15;
  int kh = (lane >> 4) << 3;                        // 0 or 8

  const float* prow = img + ((size_t)(y * 4) * IMG + (size_t)((x0 + m) * 4)) * 3;

  float4 q[6];
#pragma unroll
  for (int i = 0; i < 6; ++i) {
    static const int kbase[6] = {0, 4, 16, 20, 32, 36};
    int k = kbase[i] + kh;                          // < 48
    int ky = k / 12;
    int r = k - ky * 12;                            // 0, 4 or 8 (aligned)
    q[i] = *(const float4*)(prow + ky * (IMG * 3) + r);
  }

  v16h a0, a1;
  a0[0]  = (h16)q[0].x; a0[1]  = (h16)q[0].y; a0[2]  = (h16)q[0].z; a0[3]  = (h16)q[0].w;
  a0[4]  = (h16)q[1].x; a0[5]  = (h16)q[1].y; a0[6]  = (h16)q[1].z; a0[7]  = (h16)q[1].w;
  a0[8]  = (h16)q[2].x; a0[9]  = (h16)q[2].y; a0[10] = (h16)q[2].z; a0[11] = (h16)q[2].w;
  a0[12] = (h16)q[3].x; a0[13] = (h16)q[3].y; a0[14] = (h16)q[3].z; a0[15] = (h16)q[3].w;
  a1[0]  = (h16)q[4].x; a1[1]  = (h16)q[4].y; a1[2]  = (h16)q[4].z; a1[3]  = (h16)q[4].w;
  a1[4]  = (h16)q[5].x; a1[5]  = (h16)q[5].y; a1[6]  = (h16)q[5].z; a1[7]  = (h16)q[5].w;
#pragma unroll
  for (int e = 8; e < 16; ++e) a1[e] = (h16)0.f;   // k >= 48 pad

  const v8f zero = {0.f, 0.f, 0.f, 0.f, 0.f, 0.f, 0.f, 0.f};
#pragma unroll
  for (int t = 0; t < 8; ++t) {
    int n0 = t * 16;
    v16h b0 = frag_kmajor(WbT + n0 * 64, 64, lane);        // k 0..31
    v16h b1 = frag_kmajor(WbT + n0 * 64 + 32, 64, lane);   // k 32..63 (pad 0)
    v8f c = zero;
    c = wmma32(a0, b0, c);
    c = wmma32(a1, b1, c);
    int n = lane & 15, khr = lane >> 4;
#pragma unroll
    for (int r = 0; r < 8; ++r) {
      int mm = r + khr * 8;
      int co = n0 + n;
      feats[((y * FB) + (x0 + mm)) * C1 + co] = (h16)(c[r] + bb[co]);
    }
  }
}

// ---------------------------------------------------------------------------
// K2: inter conv 3x3, 128->256 (the 38-GFLOP GEMM).
// Input patch (3x18x128 f16) fetched by the Tensor Data Mover as a 3D tile
// (wave 0 issues, waits TENSORcnt); bias staged via async global->LDS.
// A fragments = 2x ds_load_b128, B fragments = 2x global b128 from WihT.
// ---------------------------------------------------------------------------
__global__ void __launch_bounds__(256)
k_inter(const h16* __restrict__ feats, const h16* __restrict__ WihT,
        const float* __restrict__ bi, h16* __restrict__ fmap) {
  __shared__ __align__(16) h16 patch[3 * 18 * C1];  // 13824 B
  __shared__ __align__(16) float biasS[C2];         // 1 KB

  int y = blockIdx.x >> 4;                          // 0..253
  int x0 = (blockIdx.x & 15) << 4;

  // Bias: 64 x async b128 (ASYNCcnt path)
  if (threadIdx.x < 64) {
    unsigned goff = threadIdx.x * 16;
    unsigned ldsb = (unsigned)(unsigned long long)&biasS[threadIdx.x * 4];
    asm volatile("global_load_async_to_lds_b128 %0, %1, %2"
                 :: "v"(ldsb), "v"(goff), "s"(bi) : "memory");
  }

  // Patch: one TDM 3D-tile load issued by wave 0 (TENSORcnt path)
  if ((threadIdx.x >> 5) == 0) {
#if HAVE_TDM
    unsigned long long ga =
        (unsigned long long)(const void*)(feats + (size_t)(y * FB + x0) * C1);
    unsigned ldsb = (unsigned)(unsigned long long)&patch[0];
    // D# group0: count=1 | lds_addr | global_addr[31:0] | ga[56:32]+type=2
    u32x4 g0 = {1u, ldsb, (unsigned)ga,
                (unsigned)((ga >> 32) & 0x01FFFFFFu) | (2u << 30)};
    // D# group1: data_size=2B; tensor 128 x 256 (x) ; tile 128 x 18 x 3
    //            dim0_stride=128, dim1_stride=32768 (=256*128)
    i32x8 g1 = {(int)0x00010000,          // data_size=1 (2 bytes)
                (int)(128u << 16),        // tensor_dim0 = 128
                (int)(256u << 16),        // tensor_dim1 = 256
                (int)(128u << 16),        // tile_dim0   = 128
                (int)((3u << 16) | 18u),  // tile_dim1=18, tile_dim2=3
                (int)128,                 // tensor_dim0_stride lo32
                (int)(0x8000u << 16),     // tensor_dim1_stride lo16 = 32768
                0};
    // D# group2: tensor_dim2=256 rows, dim2_stride = 256*256*128
    i32x4 g2 = {256, 0, (int)8388608, 0};
    i32x4 g3 = {0, 0, 0, 0};
#if __clang_major__ >= 23
    i32x8 g4 = {0, 0, 0, 0, 0, 0, 0, 0};
    __builtin_amdgcn_tensor_load_to_lds(g0, g1, g2, g3, g4, 0);
#else
    __builtin_amdgcn_tensor_load_to_lds(g0, g1, g2, g3, 0);
#endif
#if __has_builtin(__builtin_amdgcn_s_wait_tensorcnt)
    __builtin_amdgcn_s_wait_tensorcnt(0);
#else
    asm volatile("s_wait_tensorcnt 0" ::: "memory");
#endif
#else
    // Fallback: async global->LDS b128, lanes of wave 0
    for (int idx = threadIdx.x; idx < 864; idx += 32) {
      int c4 = idx & 15;
      int p = idx >> 4;
      int px = p % 18, py = p / 18;
      int gx = x0 + px; if (gx > FB - 1) gx = FB - 1;
      unsigned goff = (unsigned)((((y + py) * FB + gx) * C1) * 2 + c4 * 16);
      unsigned ldsb = (unsigned)(unsigned long long)&patch[idx * 8];
      asm volatile("global_load_async_to_lds_b128 %0, %1, %2"
                   :: "v"(ldsb), "v"(goff), "s"(feats) : "memory");
    }
#endif
  }
  asm volatile("s_wait_asynccnt 0" ::: "memory");
  __syncthreads();

  int lane = threadIdx.x & 31;
  int wv = threadIdx.x >> 5;                        // 0..7
  int kh = (lane >> 4) << 3;
  const v8f zero = {0.f, 0.f, 0.f, 0.f, 0.f, 0.f, 0.f, 0.f};
  v8f acc0 = zero, acc1 = zero;
  const int n0a = (wv * 2 + 0) * 16 + (lane & 15);
  const int n0b = (wv * 2 + 1) * 16 + (lane & 15);

  for (int s = 0; s < 36; ++s) {
    int kyx = s >> 2;                               // 0..8
    int ky = kyx / 3, kx = kyx % 3;
    int c0 = (s & 3) << 5;
    const h16* ap = patch + ((ky * 18) + ((lane & 15) + kx)) * C1 + c0 + kh;
    v16h a = frag_ld(ap);                           // 2x ds_load_b128
    const h16* bpa = WihT + (size_t)n0a * KI + s * 32 + kh;
    const h16* bpb = WihT + (size_t)n0b * KI + s * 32 + kh;
    __builtin_prefetch(bpa + 64, 0, 1);             // global_prefetch
    acc0 = wmma32(a, frag_ld(bpa), acc0);
    acc1 = wmma32(a, frag_ld(bpb), acc1);
  }

  int n = lane & 15, khr = lane >> 4;
#pragma unroll
  for (int r = 0; r < 8; ++r) {
    int x = x0 + r + khr * 8;
    if (x < FH) {
      int co0 = (wv * 2 + 0) * 16 + n;
      int co1 = (wv * 2 + 1) * 16 + n;
      fmap[(y * FH + x) * C2 + co0] = (h16)(acc0[r] + biasS[co0]);
      fmap[(y * FH + x) * C2 + co1] = (h16)(acc1[r] + biasS[co1]);
    }
  }
}

// ---------------------------------------------------------------------------
// K3: fused 1x1 heads (cls 9 + reg 36, N padded to 48), K=256 in 8 steps.
// Combined weights (24.6 KB) staged to LDS with async b128; B fragments via
// ds_load_b128; A fragments = 2x global b128 from fmap rows.
// ---------------------------------------------------------------------------
__global__ void __launch_bounds__(256)
k_heads(const h16* __restrict__ fmap, const h16* __restrict__ WcrT,
        const float* __restrict__ bc, const float* __restrict__ br,
        float* __restrict__ clsb, float* __restrict__ regb) {
  __shared__ __align__(16) h16 wS[NCOMB * C2];      // 24576 B

  for (int idx = threadIdx.x; idx < (NCOMB * C2) / 8; idx += 256) {  // 1536
    unsigned goff = idx * 16;
    unsigned ldsb = (unsigned)(unsigned long long)&wS[idx * 8];
    asm volatile("global_load_async_to_lds_b128 %0, %1, %2"
                 :: "v"(ldsb), "v"(goff), "s"(WcrT) : "memory");
  }
  asm volatile("s_wait_asynccnt 0" ::: "memory");
  __syncthreads();

  int lane = threadIdx.x & 31;
  int wid = blockIdx.x * 8 + (threadIdx.x >> 5);
  int pix0 = wid * 16;
  int kh = (lane >> 4) << 3;

  int pixA = pix0 + (lane & 15);
  if (pixA >= NPIX) pixA = NPIX - 1;
  const h16* arow = fmap + (size_t)pixA * C2 + kh;

  const v8f zero = {0.f, 0.f, 0.f, 0.f, 0.f, 0.f, 0.f, 0.f};
  v8f acc[3] = {zero, zero, zero};

  for (int s = 0; s < 8; ++s) {
    v16h a = frag_ld(arow + s * 32);
#pragma unroll
    for (int t = 0; t < 3; ++t) {
      const h16* bp = wS + (t * 16 + (lane & 15)) * C2 + s * 32 + kh;
      acc[t] = wmma32(a, frag_ld(bp), acc[t]);      // B via ds_load_b128
    }
  }

  int n = lane & 15, khr = lane >> 4;
#pragma unroll
  for (int t = 0; t < 3; ++t) {
    int col = t * 16 + n;
#pragma unroll
    for (int r = 0; r < 8; ++r) {
      int pix = pix0 + r + khr * 8;
      if (pix < NPIX) {
        if (col < 9)       clsb[pix * 9 + col]        = acc[t][r] + bc[col];
        else if (col < 45) regb[pix * 36 + (col - 9)] = acc[t][r] + br[col - 9];
      }
    }
  }
}

// ---------------------------------------------------------------------------
// K4: per-gt max of iou_in over all (inside) anchors -> ordered-uint atomicMax
// ---------------------------------------------------------------------------
__global__ void __launch_bounds__(256)
k_gtmax(const float* __restrict__ gt, unsigned int* __restrict__ gtkey) {
  __shared__ unsigned int smax[KGT];
  if (threadIdx.x < KGT) smax[threadIdx.x] = 0u;
  __syncthreads();

  int nidx = blockIdx.x * 256 + threadIdx.x;
  if (nidx < NANCH) {
    int q = nidx / 9, a = nidx % 9;
    int xg = q / FH, yg = q % FH;                   // anchors are (W,H,9) order
    float cx = xg * (1.0f / (float)FH);
    float cy = yg * (1.0f / (float)FH);
    float hw, hh; anchor_geom(a, hw, hh);
    float x1 = cx - hw, y1 = cy - hh, x2 = cx + hw, y2 = cy + hh;
    bool inside = (x1 >= 0.f) && (y1 >= 0.f) && (x2 < 1.f) && (y2 < 1.f);
    if (inside) {
      const float g = 1.0f / (float)IMG;
      float s1 = (x2 - x1 + g) * (y2 - y1 + g);
#pragma unroll
      for (int k = 0; k < KGT; ++k) {
        float gx1 = gt[4 * k + 0], gy1 = gt[4 * k + 1];
        float gx2 = gt[4 * k + 2], gy2 = gt[4 * k + 3];
        float s2 = (gx2 - gx1 + g) * (gy2 - gy1 + g);
        float xa = fmaxf(x1, gx1), ya = fmaxf(y1, gy1);
        float xb = fminf(x2, gx2), yb = fminf(y2, gy2);
        float inter = fmaxf(xb - xa + g, 0.f) * fmaxf(yb - ya + g, 0.f);
        float iou = inter / (s1 + s2 - inter);
        atomicMax(&smax[k], fkey(iou));
      }
    }
  }
  __syncthreads();
  if (threadIdx.x < KGT) atomicMax(&gtkey[threadIdx.x], smax[threadIdx.x]);
}

// ---------------------------------------------------------------------------
// K5: labels + BCE + smooth-L1, per-block partial sums (deterministic)
// ---------------------------------------------------------------------------
__global__ void __launch_bounds__(256)
k_loss(const float* __restrict__ gt, const unsigned int* __restrict__ gtkey,
       const float* __restrict__ clsb, const float* __restrict__ regb,
       float* __restrict__ partials) {
  __shared__ float red[4][256];
  int nidx = blockIdx.x * 256 + threadIdx.x;
  float v_bce = 0.f, v_sel = 0.f, v_box = 0.f, v_pos = 0.f;

  if (nidx < NANCH) {
    int q = nidx / 9, a = nidx % 9;
    int xg = q / FH, yg = q % FH;
    float cx = xg * (1.0f / (float)FH);
    float cy = yg * (1.0f / (float)FH);
    float hw, hh; anchor_geom(a, hw, hh);
    float x1 = cx - hw, y1 = cy - hh, x2 = cx + hw, y2 = cy + hh;
    bool inside = (x1 >= 0.f) && (y1 >= 0.f) && (x2 < 1.f) && (y2 < 1.f);

    const float g = 1.0f / (float)IMG;
    float s1 = (x2 - x1 + g) * (y2 - y1 + g);
    float max_iou = -1e30f;
    bool is_best = false;
#pragma unroll
    for (int k = 0; k < KGT; ++k) {
      float gx1 = gt[4 * k + 0], gy1 = gt[4 * k + 1];
      float gx2 = gt[4 * k + 2], gy2 = gt[4 * k + 3];
      float s2 = (gx2 - gx1 + g) * (gy2 - gy1 + g);
      float xa = fmaxf(x1, gx1), ya = fmaxf(y1, gy1);
      float xb = fminf(x2, gx2), yb = fminf(y2, gy2);
      float inter = fmaxf(xb - xa + g, 0.f) * fmaxf(yb - ya + g, 0.f);
      float iou = inter / (s1 + s2 - inter);
      max_iou = fmaxf(max_iou, iou);
      if (iou == funkey(gtkey[k])) is_best = true;
    }
    bool pos = inside && (is_best || (max_iou >= 0.5f));
    float label = inside ? 0.f : -1.f;
    if (pos) label = 1.f;
    float sel = (label != -1.f) ? 1.f : 0.f;
    float lbl = fmaxf(label, 0.f);
    float cls = clsb[nidx];                          // y-major flat == loss n
    float bce = lbl * softplusf(-cls) + (1.f - lbl) * softplusf(cls);
    v_bce = bce * sel;
    v_sel = sel;
    float m = pos ? 1.f : 0.f;
    v_pos = m;
#pragma unroll
    for (int i = 0; i < 4; ++i) {
      float ab = fabsf(regb[4 * nidx + i] * m);
      v_box += (ab < 1.f) ? (0.5f * ab * ab) : (ab - 0.5f);
    }
  }

  red[0][threadIdx.x] = v_bce;
  red[1][threadIdx.x] = v_sel;
  red[2][threadIdx.x] = v_box;
  red[3][threadIdx.x] = v_pos;
  for (int s2 = 128; s2 > 0; s2 >>= 1) {
    __syncthreads();
    if (threadIdx.x < s2)
#pragma unroll
      for (int j = 0; j < 4; ++j)
        red[j][threadIdx.x] += red[j][threadIdx.x + s2];
  }
  if (threadIdx.x == 0)
#pragma unroll
    for (int j = 0; j < 4; ++j)
      partials[blockIdx.x * 4 + j] = red[j][0];
}

// ---------------------------------------------------------------------------
// K6: final reduce + combine losses
// ---------------------------------------------------------------------------
__global__ void __launch_bounds__(256)
k_final(const float* __restrict__ partials, int nblk, float* __restrict__ out) {
  __shared__ float red[4][256];
  float s[4] = {0.f, 0.f, 0.f, 0.f};
  for (int i = threadIdx.x; i < nblk; i += 256)
#pragma unroll
    for (int j = 0; j < 4; ++j) s[j] += partials[i * 4 + j];
#pragma unroll
  for (int j = 0; j < 4; ++j) red[j][threadIdx.x] = s[j];
  for (int s2 = 128; s2 > 0; s2 >>= 1) {
    __syncthreads();
    if (threadIdx.x < s2)
#pragma unroll
      for (int j = 0; j < 4; ++j)
        red[j][threadIdx.x] += red[j][threadIdx.x + s2];
  }
  if (threadIdx.x == 0)
    out[0] = red[0][0] / red[1][0] + red[2][0] / red[3][0];
}

// ---------------------------------------------------------------------------
// Host launcher
// ---------------------------------------------------------------------------
extern "C" void kernel_launch(void* const* d_in, const int* in_sizes, int n_in,
                              void* d_out, int out_size, void* d_ws, size_t ws_size,
                              hipStream_t stream) {
  const float* img = (const float*)d_in[0];
  const float* gt  = (const float*)d_in[1];
  const float* Wb  = (const float*)d_in[2];
  const float* bb  = (const float*)d_in[3];
  const float* Wi  = (const float*)d_in[4];
  const float* bi  = (const float*)d_in[5];
  const float* Wc  = (const float*)d_in[6];
  const float* bc  = (const float*)d_in[7];
  const float* Wr  = (const float*)d_in[8];
  const float* br  = (const float*)d_in[9];

  char* w = (char*)d_ws;
  auto carve = [&](size_t bytes) -> void* {
    void* p = (void*)w;
    w += (bytes + 255) & ~(size_t)255;
    return p;
  };
  h16*  feats = (h16*)carve((size_t)FB * FB * C1 * sizeof(h16));      // 16.8 MB
  h16*  fmap  = (h16*)carve((size_t)NPIX * C2 * sizeof(h16));         // 33.0 MB
  h16*  WbT   = (h16*)carve(8192 * sizeof(h16));
  h16*  WihT  = (h16*)carve((size_t)C2 * KI * sizeof(h16));
  h16*  WcrT  = (h16*)carve((size_t)NCOMB * C2 * sizeof(h16));
  float* clsb = (float*)carve((size_t)NANCH * sizeof(float));
  float* regb = (float*)carve((size_t)NANCH * 4 * sizeof(float));
  unsigned int* gtkey = (unsigned int*)carve(KGT * sizeof(unsigned int));
  float* partials = (float*)carve((size_t)ABLK * 4 * sizeof(float));

  // tasks: 8192 + 294912 + 12288 + 16 = 315408 -> 1233 blocks
  k_prep<<<1233, 256, 0, stream>>>(Wb, Wi, Wc, Wr, WbT, WihT, WcrT, gtkey);
  k_backbone<<<512, 256, 0, stream>>>(img, WbT, bb, feats);
  k_inter<<<FH * 16, 256, 0, stream>>>(feats, WihT, bi, fmap);
  k_heads<<<505, 256, 0, stream>>>(fmap, WcrT, bc, br, clsb, regb);
  k_gtmax<<<ABLK, 256, 0, stream>>>(gt, gtkey);
  k_loss<<<ABLK, 256, 0, stream>>>(gt, gtkey, clsb, regb, partials);
  k_final<<<1, 256, 0, stream>>>(partials, ABLK, (float*)d_out);
}
